// RATEAttention_68925635166857
// MI455X (gfx1250) — matrix operand order repaired and checked by reading, not compiled, not run on hardware
//
#include <hip/hip_runtime.h>
#include <hip/hip_bf16.h>

// ---------------------------------------------------------------------------
// RATE attention for MI455X (gfx1250), bf16 WMMA path + async-LDS weight
// staging. D=1024, H=16, dk=64, B=2, L=2048.
// d_out = [ out f32 (2*2048*1024) | attn_mean f32 (2*2048*2048) ]
// ---------------------------------------------------------------------------

#define B_   2
#define L_   2048
#define D_   1024
#define H_   16
#define DK_  64
#define ML_  (B_ * L_)          // 4096 rows

#define KC_     64              // K-chunk staged in LDS per iteration
#define TPITCH_ 72              // padded LDS tile row pitch (bf16 elems): 144B
                                // -> lane n reads at dword stride 36 (mod 64
                                //    distinct for n=0..15: conflict-free)

typedef __attribute__((ext_vector_type(16))) __bf16 v16bf;
typedef __attribute__((ext_vector_type(8)))  __bf16 v8bf;
typedef __attribute__((ext_vector_type(8)))  float  v8f;

union BF16x16 { v16bf v; v8bf h[2]; };

// Workspace layout (bytes). Requires ~56.1 MB of d_ws.
static const size_t OFF_WQT  = 0;
static const size_t OFF_WKT  = OFF_WQT  + (size_t)D_ * D_ * 2;
static const size_t OFF_WVT  = OFF_WKT  + (size_t)D_ * D_ * 2;
static const size_t OFF_WOT  = OFF_WVT  + (size_t)D_ * D_ * 2;
static const size_t OFF_XBF  = OFF_WOT  + (size_t)D_ * D_ * 2;
static const size_t OFF_XVBF = OFF_XBF  + (size_t)ML_ * D_ * 2;
static const size_t OFF_Q    = OFF_XVBF + (size_t)ML_ * D_ * 2;
static const size_t OFF_K    = OFF_Q    + (size_t)ML_ * D_ * 2;
static const size_t OFF_VT   = OFF_K    + (size_t)ML_ * D_ * 2;
static const size_t OFF_CTX  = OFF_VT   + (size_t)ML_ * D_ * 2;
static const size_t OFF_REL  = OFF_CTX  + (size_t)ML_ * D_ * 2;

// A-operand (16x32 bf16) fetch from a K-contiguous bf16 row.
// lane<16: K = kbase+{0..15}; lane>=16: K = kbase+{16..31}.
__device__ __forceinline__ v16bf load_a_row(const __bf16* row, int kbase, int half) {
  BF16x16 u;
  u.h[0] = *(const v8bf*)(row + kbase + half * 16);
  u.h[1] = *(const v8bf*)(row + kbase + half * 16 + 8);
  return u.v;
}

// 32B B-operand gather from LDS (16B-aligned halves).
__device__ __forceinline__ v16bf load_b_lds(const __bf16* p) {
  BF16x16 u;
  u.h[0] = *(const v8bf*)(p);
  u.h[1] = *(const v8bf*)(p + 8);
  return u.v;
}

// Async global->LDS 16B copy (ASYNCcnt path). Low 32 bits of a flat LDS
// pointer are the wave-relative LDS byte address (aperture truncation rule).
__device__ __forceinline__ void async_copy_b128(const __bf16* gptr, __bf16* lptr) {
  uint32_t l = (uint32_t)(uintptr_t)lptr;
  uint64_t g = (uint64_t)(uintptr_t)gptr;
  asm volatile("global_load_async_to_lds_b128 %0, %1, off"
               :: "v"(l), "v"(g) : "memory");
}
__device__ __forceinline__ void wait_async0() {
  asm volatile("s_wait_asynccnt 0x0" ::: "memory");
}

// ---------------------------------------------------------------------------
// Prep: transpose weights to bf16 (Wt[j][d] = W[d][j]) so B operand rows are
// K-contiguous.
// ---------------------------------------------------------------------------
__global__ __launch_bounds__(256) void prep_weights(
    const float* __restrict__ Wq, const float* __restrict__ Wk,
    const float* __restrict__ Wv, const float* __restrict__ Wo,
    __bf16* __restrict__ wqt, __bf16* __restrict__ wkt,
    __bf16* __restrict__ wvt, __bf16* __restrict__ wot)
{
  int idx = blockIdx.x * 256 + threadIdx.x;           // 0 .. 4*1024*1024-1
  int w = idx >> 20;
  int r = idx & ((1 << 20) - 1);
  int j = r >> 10;          // output feature (row of Wt)
  int d = r & 1023;         // reduction index (contiguous in Wt row)
  const float* W = (w == 0) ? Wq : (w == 1) ? Wk : (w == 2) ? Wv : Wo;
  __bf16* Wt     = (w == 0) ? wqt : (w == 1) ? wkt : (w == 2) ? wvt : wot;
  Wt[(size_t)j * D_ + d] = (__bf16)W[(size_t)d * D_ + j];
}

// ---------------------------------------------------------------------------
// Prep: x -> bf16, x*delta -> bf16, rel_bias = log(delta + 1e-6)
// ---------------------------------------------------------------------------
__global__ __launch_bounds__(256) void prep_x(
    const float* __restrict__ x, const float* __restrict__ delta,
    __bf16* __restrict__ xbf, __bf16* __restrict__ xvbf,
    float* __restrict__ rel)
{
  int idx = blockIdx.x * 256 + threadIdx.x;           // 0 .. ML_*D_-1
  int row = idx >> 10;
  float xv = x[idx];
  xbf[idx]  = (__bf16)xv;
  xvbf[idx] = (__bf16)(xv * delta[row]);
  if (idx < B_ * L_) rel[idx] = __logf(delta[idx] + 1e-6f);
}

// ---------------------------------------------------------------------------
// Shared GEMM core: 4 waves x (16M x 64N) tiles, weight chunk (64N x 64K)
// double-buffered in LDS via async loads, A streamed from global.
// Accumulators returned in acc[4] (v8f each).
// ---------------------------------------------------------------------------
#define GEMM_BODY(Asrc, Wt)                                                    \
  __shared__ __bf16 wtile[2][64 * TPITCH_];                                    \
  const int wave   = threadIdx.x >> 5;                                         \
  const int lane   = threadIdx.x & 31;                                         \
  const int lane15 = lane & 15;                                                \
  const int half   = lane >> 4;                                                \
  const int mt = blockIdx.x * 4 + wave;                                        \
  const int nb = blockIdx.y * 64;                                              \
  const __bf16* arow = (Asrc) + (size_t)(mt * 16 + lane15) * D_;               \
  v8f acc[4] = {{}, {}, {}, {}};                                               \
  /* prologue: stage chunk 0 */                                                \
  for (int i = 0; i < 4; ++i) {                                                \
    int c = threadIdx.x + 128 * i;   /* 0..511 */                              \
    int n = c >> 3, k8 = c & 7;                                                \
    async_copy_b128((Wt) + (size_t)(nb + n) * D_ + k8 * 8,                     \
                    &wtile[0][n * TPITCH_ + k8 * 8]);                          \
  }                                                                            \
  wait_async0();                                                               \
  __syncthreads();                                                             \
  for (int kc = 0; kc < D_; kc += KC_) {                                       \
    int bb = (kc / KC_) & 1;                                                   \
    if (kc + KC_ < D_) {                                                       \
      for (int i = 0; i < 4; ++i) {                                            \
        int c = threadIdx.x + 128 * i;                                         \
        int n = c >> 3, k8 = c & 7;                                            \
        async_copy_b128((Wt) + (size_t)(nb + n) * D_ + (kc + KC_) + k8 * 8,    \
                        &wtile[bb ^ 1][n * TPITCH_ + k8 * 8]);                 \
      }                                                                        \
    }                                                                          \
    _Pragma("unroll")                                                          \
    for (int ks = 0; ks < KC_; ks += 32) {                                     \
      v16bf a = load_a_row(arow, kc + ks, half);                               \
      _Pragma("unroll")                                                        \
      for (int t = 0; t < 4; ++t) {                                            \
        const __bf16* bp =                                                     \
            &wtile[bb][(t * 16 + lane15) * TPITCH_ + ks + half * 16];          \
        v16bf bt = load_b_lds(bp);                                             \
        acc[t] = __builtin_amdgcn_wmma_f32_16x16x32_bf16(                      \
            false, a, false, bt, (short)0, acc[t], false, false);              \
      }                                                                        \
    }                                                                          \
    wait_async0();                                                             \
    __syncthreads();                                                           \
  }

// ---------------------------------------------------------------------------
// Fused Q/K/V projection. blockIdx.z selects which projection.
// Q,K stored as [b,h,l,dk]; V stored transposed [b,h,dk,l].
// ---------------------------------------------------------------------------
__global__ __launch_bounds__(128) void proj_qkv(
    const __bf16* __restrict__ xbf, const __bf16* __restrict__ xvbf,
    const __bf16* __restrict__ wqt, const __bf16* __restrict__ wkt,
    const __bf16* __restrict__ wvt,
    const float* __restrict__ bq, const float* __restrict__ bk,
    const float* __restrict__ bv,
    __bf16* __restrict__ qout, __bf16* __restrict__ kout,
    __bf16* __restrict__ vtout)
{
  const int z = blockIdx.z;                 // 0=Q 1=K 2=V
  const __bf16* A   = (z == 2) ? xvbf : xbf;
  const __bf16* W   = (z == 0) ? wqt : (z == 1) ? wkt : wvt;
  const float* bias = (z == 0) ? bq  : (z == 1) ? bk  : bv;

  GEMM_BODY(A, W)

#pragma unroll
  for (int t = 0; t < 4; ++t) {
    int n  = nb + t * 16 + lane15;          // output feature
    int hh = n >> 6;
    int jd = n & 63;
    float bs = bias[n];
#pragma unroll
    for (int r = 0; r < 8; ++r) {
      int m   = r + half * 8;
      int row = mt * 16 + m;
      int b   = row >> 11;
      int l   = row & (L_ - 1);
      __bf16 val = (__bf16)(acc[t][r] + bs);
      if (z == 0)      qout [((size_t)(b * H_ + hh) * L_ + l) * DK_ + jd] = val;
      else if (z == 1) kout [((size_t)(b * H_ + hh) * L_ + l) * DK_ + jd] = val;
      else             vtout[((size_t)(b * H_ + hh) * DK_ + jd) * L_ + l] = val;
    }
  }
}

// ---------------------------------------------------------------------------
// Output projection: out = ctx @ Wo + bo  (f32 out to d_out).
// ---------------------------------------------------------------------------
__global__ __launch_bounds__(128) void proj_out(
    const __bf16* __restrict__ ctx, const __bf16* __restrict__ wot,
    const float* __restrict__ bo, float* __restrict__ out)
{
  GEMM_BODY(ctx, wot)

#pragma unroll
  for (int t = 0; t < 4; ++t) {
    int n = nb + t * 16 + lane15;
    float bs = bo[n];
#pragma unroll
    for (int r = 0; r < 8; ++r) {
      int m   = r + half * 8;
      int row = mt * 16 + m;
      out[(size_t)row * D_ + n] = acc[t][r] + bs;
    }
  }
}

// ---------------------------------------------------------------------------
// Attention: one block per (b, 16-query tile); heads looped in-block so the
// attn_mean accumulation is a deterministic RMW. Dynamic LDS holds the full
// 16 x 2048 f32 score row (128 KB of the WGP's 320 KB).
// ---------------------------------------------------------------------------
__global__ __launch_bounds__(256) void attn_kernel(
    const __bf16* __restrict__ qbf, const __bf16* __restrict__ kbf,
    const __bf16* __restrict__ vtbf, const float* __restrict__ rel,
    __bf16* __restrict__ ctx, float* __restrict__ attn_mean)
{
  extern __shared__ float scores[];         // 16*2048 f32
  __shared__ float red[16][16];
  __shared__ float rowred[16];
  __shared__ float O_lds[16 * 64];

  const int tid   = threadIdx.x;
  const int wave  = tid >> 5;
  const int lane  = tid & 31;
  const int lane15 = lane & 15;
  const int half   = lane >> 4;
  const int qt = blockIdx.x;                // 0..127
  const int b  = blockIdx.y;                // 0..1
  const int q0 = qt * 16;
  const int sr = tid >> 4;                  // softmax row
  const int sc = tid & 15;                  // softmax column lane

  for (int e = tid; e < 16 * 64; e += 256) O_lds[e] = 0.f;

  for (int h = 0; h < H_; ++h) {
    const __bf16* qbase = qbf  + (size_t)(b * H_ + h) * L_ * DK_;
    const __bf16* kbase = kbf  + (size_t)(b * H_ + h) * L_ * DK_;
    const __bf16* vbase = vtbf + (size_t)(b * H_ + h) * DK_ * L_;

    // ---- scores = Q K^T / 8 - 0.1*|q-kv| + rel[kv] -----------------------
    const __bf16* qrow = qbase + (size_t)(q0 + lane15) * DK_;
    v16bf aq0 = load_a_row(qrow, 0,  half);
    v16bf aq1 = load_a_row(qrow, 32, half);

    for (int t = wave; t < L_ / 16; t += 8) {
      int kv0 = t * 16;
      const __bf16* krow = kbase + (size_t)(kv0 + lane15) * DK_ + half * 16;
      v16bf b0 = *(const v16bf*)(krow);
      v16bf b1 = *(const v16bf*)(krow + 32);
      v8f s = {};
      s = __builtin_amdgcn_wmma_f32_16x16x32_bf16(false, aq0, false, b0,
                                                  (short)0, s, false, false);
      s = __builtin_amdgcn_wmma_f32_16x16x32_bf16(false, aq1, false, b1,
                                                  (short)0, s, false, false);
      int kv = kv0 + lane15;
      float rb = rel[b * L_ + kv];
#pragma unroll
      for (int r = 0; r < 8; ++r) {
        int m = r + half * 8;
        int q = q0 + m;
        scores[m * L_ + kv] =
            s[r] * 0.125f - 0.1f * fabsf((float)(q - kv)) + rb;
      }
    }
    __syncthreads();

    // ---- softmax over kv (row of 2048), 16 threads per row ---------------
    float mx = -3.0e38f;
    for (int i = 0; i < 128; ++i)
      mx = fmaxf(mx, scores[sr * L_ + sc + 16 * i]);
    red[sr][sc] = mx;
    __syncthreads();
    if (tid < 16) {
      float m2 = red[tid][0];
      for (int i = 1; i < 16; ++i) m2 = fmaxf(m2, red[tid][i]);
      rowred[tid] = m2;
    }
    __syncthreads();
    float rm  = rowred[sr];
    float sum = 0.f;
    for (int i = 0; i < 128; ++i) {
      int idx = sr * L_ + sc + 16 * i;
      float e = __expf(scores[idx] - rm);
      scores[idx] = e;
      sum += e;
    }
    red[sr][sc] = sum;
    __syncthreads();
    if (tid < 16) {
      float s2 = 0.f;
      for (int i = 0; i < 16; ++i) s2 += red[tid][i];
      rowred[tid] = 1.0f / s2;
    }
    __syncthreads();
    float inv = rowred[sr];
    float* amrow = attn_mean + (size_t)(b * L_ + q0 + sr) * L_;
    for (int i = 0; i < 128; ++i) {
      int c = sc + 16 * i;
      float p = scores[sr * L_ + c] * inv;
      scores[sr * L_ + c] = p;
      float prev = (h == 0) ? 0.f : amrow[c];
      amrow[c] = prev + p * (1.0f / H_);
    }
    __syncthreads();

    // ---- context: O = P @ V (kv split across 8 waves, LDS reduce) --------
    v8f o[4] = {{}, {}, {}, {}};
    for (int j = 0; j < 8; ++j) {
      int kv0 = (wave + 8 * j) * 32;
      BF16x16 ua;
      const float* prow = scores + lane15 * L_ + kv0;
#pragma unroll
      for (int i = 0; i < 8; ++i) ua.v[i]     = (__bf16)prow[half * 16 + i];
#pragma unroll
      for (int i = 0; i < 8; ++i) ua.v[8 + i] = (__bf16)prow[half * 16 + 8 + i];
#pragma unroll
      for (int t = 0; t < 4; ++t) {
        int d = t * 16 + lane15;
        v16bf bv = *(const v16bf*)(vbase + (size_t)d * L_ + kv0 + half * 16);
        o[t] = __builtin_amdgcn_wmma_f32_16x16x32_bf16(
            false, ua.v, false, bv, (short)0, o[t], false, false);
      }
    }
#pragma unroll
    for (int t = 0; t < 4; ++t) {
      int d = t * 16 + lane15;
#pragma unroll
      for (int r = 0; r < 8; ++r) {
        int m = r + half * 8;
        atomicAdd(&O_lds[m * 64 + d], o[t][r]);   // ds_add_f32
      }
    }
    __syncthreads();

    // ---- emit context tile (bf16), clear O for next head -----------------
    for (int e = tid; e < 16 * 64; e += 256) {
      int m = e >> 6, d = e & 63;
      ctx[(size_t)(b * L_ + q0 + m) * D_ + h * DK_ + d] = (__bf16)O_lds[e];
    }
    __syncthreads();
    for (int e = tid; e < 16 * 64; e += 256) O_lds[e] = 0.f;
  }
}

// ---------------------------------------------------------------------------
extern "C" void kernel_launch(void* const* d_in, const int* in_sizes, int n_in,
                              void* d_out, int out_size, void* d_ws,
                              size_t ws_size, hipStream_t stream)
{
  const float* x     = (const float*)d_in[0];
  const float* delta = (const float*)d_in[1];
  const float* Wq    = (const float*)d_in[2];
  const float* bq    = (const float*)d_in[3];
  const float* Wk    = (const float*)d_in[4];
  const float* bk    = (const float*)d_in[5];
  const float* Wv    = (const float*)d_in[6];
  const float* bv    = (const float*)d_in[7];
  const float* Wo    = (const float*)d_in[8];
  const float* bo    = (const float*)d_in[9];

  char* ws = (char*)d_ws;
  __bf16* wqt  = (__bf16*)(ws + OFF_WQT);
  __bf16* wkt  = (__bf16*)(ws + OFF_WKT);
  __bf16* wvt  = (__bf16*)(ws + OFF_WVT);
  __bf16* wot  = (__bf16*)(ws + OFF_WOT);
  __bf16* xbf  = (__bf16*)(ws + OFF_XBF);
  __bf16* xvbf = (__bf16*)(ws + OFF_XVBF);
  __bf16* qb   = (__bf16*)(ws + OFF_Q);
  __bf16* kb   = (__bf16*)(ws + OFF_K);
  __bf16* vtb  = (__bf16*)(ws + OFF_VT);
  __bf16* ctx  = (__bf16*)(ws + OFF_CTX);
  float*  relb = (float*) (ws + OFF_REL);

  float* out       = (float*)d_out;
  float* attn_mean = out + (size_t)ML_ * D_;

  // Raise dynamic-LDS cap for the attention kernel (capture-safe host call).
  (void)hipFuncSetAttribute((const void*)attn_kernel,
                            hipFuncAttributeMaxDynamicSharedMemorySize,
                            16 * L_ * (int)sizeof(float));

  prep_weights<<<(4 * D_ * D_) / 256, 256, 0, stream>>>(
      Wq, Wk, Wv, Wo, wqt, wkt, wvt, wot);
  prep_x<<<(ML_ * D_) / 256, 256, 0, stream>>>(x, delta, xbf, xvbf, relb);
  proj_qkv<<<dim3(ML_ / 64, D_ / 64, 3), 128, 0, stream>>>(
      xbf, xvbf, wqt, wkt, wvt, bq, bk, bv, qb, kb, vtb);
  attn_kernel<<<dim3(L_ / 16, B_, 1), 256, 16 * L_ * sizeof(float), stream>>>(
      qb, kb, vtb, relb, ctx, attn_mean);
  proj_out<<<dim3(ML_ / 64, D_ / 64, 1), 128, 0, stream>>>(ctx, wot, bo, out);
}
